// BDEC_67877663146483
// MI455X (gfx1250) — matrix operand order repaired and checked by reading, not compiled, not run on hardware
//
#include <hip/hip_runtime.h>
#include <math.h>

// ---------------- WMMA types ----------------
typedef _Float16 f16_t;
typedef __attribute__((ext_vector_type(16))) _Float16 v16h;
typedef __attribute__((ext_vector_type(8)))  float    v8f;

#define N_PTS 29696
#define KC    100
#define NZD   32
#define EPSV  1e-5f
#define VDOF  1.0f
#define CCONST 1.0f

// ---- CDNA5 async global->LDS path (probe via __has_builtin; safe fallback) ----
#if defined(__has_builtin)
#  if __has_builtin(__builtin_amdgcn_global_load_async_to_lds_b128) && \
      __has_builtin(__builtin_amdgcn_s_wait_asynccnt)
#    define HAVE_ASYNC 1
#  endif
#endif
#ifndef HAVE_ASYNC
#  define HAVE_ASYNC 0
#endif

#if HAVE_ASYNC
typedef __attribute__((__vector_size__(16))) int v4i;
typedef __attribute__((address_space(1))) v4i as1_v4i;   // global
typedef __attribute__((address_space(3))) v4i as3_v4i;   // LDS
#endif

// ---------------- GEMM: C = act(A[NxK] @ Wt[MxK]^T + bias) ----------------
// A, Wt are f16 (prepped); C stored f16 (MODE 0 also stores f32 for clustering).
// MODE: 0 = store f32+f16 (z), 1 = relu + store f16, 2 = sum((C-X)^2) -> red_acc
// GUARD: epilogue column bounds checks (Mout=32 case; B reads use zero-padded Wt)
#define BM 128
#define BN 128
#define BK 32
#define LDA 40   // BK + 8 f16 padding (conflict-free ds_load_b128)
#define LDB 40

template <int MODE, bool GUARD>
__global__ __launch_bounds__(256) void gemm_kernel(
    const f16_t* __restrict__ A, const f16_t* __restrict__ Wt,
    const float* __restrict__ bias, f16_t* __restrict__ C16,
    float* __restrict__ C32, const float* __restrict__ X,
    float* __restrict__ red_acc, int K, int Mout)
{
    __shared__ f16_t As[2][BM * LDA];
    __shared__ f16_t Bs[2][BN * LDB];
    __shared__ float red_lds[256];

    const int tid    = threadIdx.x;
    const int lane   = tid & 31;
    const int wid    = tid >> 5;
    const int wave_m = wid & 3;   // 0..3 -> 32-row slabs
    const int wave_n = wid >> 2;  // 0..1 -> 64-col slabs
    const int m0 = blockIdx.x * BM;
    const int n0 = blockIdx.y * BN;

    v8f acc[2][4];
#pragma unroll
    for (int tm = 0; tm < 2; ++tm)
#pragma unroll
        for (int tn = 0; tn < 4; ++tn)
#pragma unroll
            for (int v = 0; v < 8; ++v) acc[tm][tn][v] = 0.0f;

#if HAVE_ASYNC
    // pure DMA staging: per-lane 16B chunks, A tile 128x32 f16, B tile 128x32 f16
    auto stage = [&](int kk, int b) {
#pragma unroll
        for (int i = 0; i < 2; ++i) {
            int ca  = tid + i * 256;          // 0..511 chunks
            int row = ca >> 2;                // 0..127
            int ch  = (ca & 3) << 3;          // f16 offset {0,8,16,24}
            __builtin_amdgcn_global_load_async_to_lds_b128(
                (as1_v4i*)(void*)(A + (size_t)(m0 + row) * K + kk + ch),
                (as3_v4i*)(void*)&As[b][row * LDA + ch], 0, 0);
            __builtin_amdgcn_global_load_async_to_lds_b128(
                (as1_v4i*)(void*)(Wt + (size_t)(n0 + row) * K + kk + ch),
                (as3_v4i*)(void*)&Bs[b][row * LDB + ch], 0, 0);
        }
    };
#else
    uint4 rca[2], rcb[2];
    auto loadRegs = [&](int kk) {
#pragma unroll
        for (int i = 0; i < 2; ++i) {
            int ca = tid + i * 256, row = ca >> 2, ch = (ca & 3) << 3;
            rca[i] = *(const uint4*)(A  + (size_t)(m0 + row) * K + kk + ch);
            rcb[i] = *(const uint4*)(Wt + (size_t)(n0 + row) * K + kk + ch);
        }
    };
    auto storeRegs = [&](int b) {
#pragma unroll
        for (int i = 0; i < 2; ++i) {
            int ca = tid + i * 256, row = ca >> 2, ch = (ca & 3) << 3;
            *(uint4*)&As[b][row * LDA + ch] = rca[i];
            *(uint4*)&Bs[b][row * LDB + ch] = rcb[i];
        }
    };
#endif

    auto compute = [&](int b) {
        v16h af[2], bf[4];
        const int kgo = (lane < 16) ? 0 : 8;      // A: lanes>=16 take K=8..15 / 24..31
#pragma unroll
        for (int tm = 0; tm < 2; ++tm) {
            int row = wave_m * 32 + tm * 16 + (lane & 15);
            const f16_t* pa = &As[b][row * LDA];
#pragma unroll
            for (int i = 0; i < 8; ++i) af[tm][i] = pa[kgo + i];
#pragma unroll
            for (int i = 0; i < 8; ++i) af[tm][8 + i] = pa[kgo + 16 + i];
        }
        const int kgb = (lane < 16) ? 0 : 8;      // B: lanes>=16 take K=8..15 / 24..31
#pragma unroll
        for (int tn = 0; tn < 4; ++tn) {
            int col = wave_n * 64 + tn * 16 + (lane & 15);
            const f16_t* pb = &Bs[b][col * LDB + kgb];
#pragma unroll
            for (int i = 0; i < 16; ++i) bf[tn][i] = pb[i + ((i >= 8) ? 8 : 0)];
        }
#pragma unroll
        for (int tm = 0; tm < 2; ++tm)
#pragma unroll
            for (int tn = 0; tn < 4; ++tn)
                acc[tm][tn] = __builtin_amdgcn_wmma_f32_16x16x32_f16(
                    false, af[tm], false, bf[tn],
                    (short)0, acc[tm][tn], false, false);
    };

    // ---- pipelined main loop (double-buffered LDS) ----
#if HAVE_ASYNC
    stage(0, 0);
    __builtin_amdgcn_s_wait_asynccnt(0);
    __syncthreads();
    int p = 0;
    for (int k0 = 0;;) {
        const int k1 = k0 + BK;
        const bool more = (k1 < K);
        if (more) stage(k1, p ^ 1);   // DMA next tile while computing this one
        compute(p);
        if (!more) break;
        __builtin_amdgcn_s_wait_asynccnt(0);
        __syncthreads();
        p ^= 1;
        k0 = k1;
    }
#else
    loadRegs(0);
    storeRegs(0);
    __syncthreads();
    int p = 0;
    for (int k0 = 0;;) {
        const int k1 = k0 + BK;
        const bool more = (k1 < K);
        if (more) loadRegs(k1);
        compute(p);
        if (!more) break;
        storeRegs(p ^ 1);
        __syncthreads();
        p ^= 1;
        k0 = k1;
    }
#endif

    // ---- epilogue ----
    float local_red = 0.0f;
#pragma unroll
    for (int tm = 0; tm < 2; ++tm) {
#pragma unroll
        for (int tn = 0; tn < 4; ++tn) {
            int row_b = m0 + wave_m * 32 + tm * 16 + ((lane < 16) ? 0 : 8);
            int col   = n0 + wave_n * 64 + tn * 16 + (lane & 15);
            bool cok  = !GUARD || (col < Mout);
            float b   = cok ? bias[col] : 0.0f;
#pragma unroll
            for (int v = 0; v < 8; ++v) {
                int row = row_b + v;
                float c = acc[tm][tn][v] + b;
                if (MODE == 1) c = fmaxf(c, 0.0f);
                if (MODE == 2) {
                    if (cok) {
                        float d = c - X[(size_t)row * Mout + col];
                        local_red += d * d;
                    }
                } else {
                    if (cok) {
                        C16[(size_t)row * Mout + col] = (f16_t)c;
                        if (MODE == 0) C32[(size_t)row * Mout + col] = c;
                    }
                }
            }
        }
    }
    if (MODE == 2) {
        red_lds[tid] = local_red;
        __syncthreads();
        for (int s = 128; s > 0; s >>= 1) {
            if (tid < s) red_lds[tid] += red_lds[tid + s];
            __syncthreads();
        }
        if (tid == 0) atomicAdd(red_acc, red_lds[0]);
    }
}

// ---------------- prep: f32 -> f16, and weight transpose+convert ----------------
__global__ void cvt_f16_kernel(const float* __restrict__ in, f16_t* __restrict__ out, int n) {
    int i = blockIdx.x * 256 + threadIdx.x;
    if (i < n) out[i] = (f16_t)in[i];
}
// Wt[m][k] = (f16)W[k][m]; rows m in [M, Mpad) zero-padded
__global__ void transpose_cvt_kernel(const float* __restrict__ W, f16_t* __restrict__ Wt,
                                     int K, int M, int Mpad) {
    __shared__ float t[32][33];
    int kb = blockIdx.x * 32, mb = blockIdx.y * 32;
    int tx = threadIdx.x, ty = threadIdx.y;   // 32 x 8
    for (int r = 0; r < 32; r += 8) {
        int k = kb + ty + r, m = mb + tx;
        t[ty + r][tx] = (k < K && m < M) ? W[(size_t)k * M + m] : 0.0f;
    }
    __syncthreads();
    for (int r = 0; r < 32; r += 8) {
        int m = mb + ty + r, k = kb + tx;
        if (m < Mpad && k < K) Wt[(size_t)m * K + k] = (f16_t)t[tx][ty + r];
    }
}

// ---------------- clustering: one wave per point ----------------
__global__ __launch_bounds__(256) void cluster_kernel(
    const float* __restrict__ Z, const float* __restrict__ T1,
    const float* __restrict__ CL, float* __restrict__ Qm,
    int* __restrict__ PRED, float* __restrict__ u_glob,
    int* __restrict__ cnt, float* __restrict__ S_glob, int Npts)
{
    __shared__ float cl[KC * NZD];
    __shared__ float zp_l[8][NZD];
    __shared__ float u_lds[KC];

    const int tid = threadIdx.x, lane = tid & 31, wid = tid >> 5;
    for (int i = tid; i < KC * NZD; i += 256) cl[i] = CL[i];
    for (int i = tid; i < KC; i += 256) u_lds[i] = 0.0f;
    __syncthreads();

    float s_total = 0.0f;
    for (int base = blockIdx.x * 8; base < Npts; base += gridDim.x * 8) {
        const int pt = base + wid;   // Npts % 8 == 0 -> always valid
        float zv = Z[(size_t)pt * NZD + lane];
        float t0 = T1[pt * 3 + 0] * 0.01f;
        float ta = T1[pt * 3 + 1] * 0.01f;
        float tb = T1[pt * 3 + 2] * 0.01f;
        float mult = t0 * ta * tb * 0.99f + 1.0f;
        float zp = zv * mult;

        float s = zp;
        for (int off = 16; off; off >>= 1) s += __shfl_xor(s, off, 32);
        float mean = s * (1.0f / 32.0f);
        float d = zp - mean;
        float vs = d * d;
        for (int off = 16; off; off >>= 1) vs += __shfl_xor(vs, off, 32);
        float stdv = sqrtf(vs * (1.0f / 31.0f));   // ddof=1
        float zpn = d / stdv;
        zp_l[wid][lane] = zpn;
        __builtin_amdgcn_wave_barrier();

        float qv[4];
        float qs = 0.0f;
#pragma unroll
        for (int r = 0; r < 4; ++r) {
            int c = lane + 32 * r;
            float t = 0.0f;
            if (c < KC) {
                float dist = 0.0f;
#pragma unroll
                for (int d2 = 0; d2 < NZD; ++d2) {
                    float df = zp_l[wid][d2] - cl[c * NZD + d2];
                    dist += df * df;
                }
                // q = (1/(eps+dist/v))^(-(v+1)/2) == (eps+dist/v)^((v+1)/2); v=1 -> identity
                t = EPSV + dist / VDOF;
            }
            qv[r] = t;
            qs += t;
        }
        for (int off = 16; off; off >>= 1) qs += __shfl_xor(qs, off, 32);
        float inv = 1.0f / qs;

        float bv = -1.0f; int bi = 0;
#pragma unroll
        for (int r = 0; r < 4; ++r) {
            int c = lane + 32 * r;
            if (c < KC) {
                float qn = qv[r] * inv;
                Qm[(size_t)pt * KC + c] = qn;
                atomicAdd(&u_lds[c], qn);
                float lg = logf(qn);
                float om = 1.0f - qn;
                float tq = om * om * lg * (float)Npts;
                s_total += sqrtf(-1.0f / tq);
                if (qv[r] > bv) { bv = qv[r]; bi = c; }   // first index on ties
            }
        }
        for (int off = 16; off; off >>= 1) {
            float ov = __shfl_xor(bv, off, 32);
            int   oi = __shfl_xor(bi, off, 32);
            if (ov > bv || (ov == bv && oi < bi)) { bv = ov; bi = oi; }
        }
        if (lane == 0) {
            PRED[pt] = bi;
            atomicAdd(&cnt[bi], 1);
        }
    }
    for (int off = 16; off; off >>= 1) s_total += __shfl_xor(s_total, off, 32);
    if (lane == 0) atomicAdd(S_glob, s_total);
    __syncthreads();
    for (int i = tid; i < KC; i += 256) atomicAdd(&u_glob[i], u_lds[i]);
}

// ---------------- finalize u, v_vec, f and cluster-pair distance ----------------
__global__ __launch_bounds__(128) void finalize_kernel(
    const float* __restrict__ u_raw, const int* __restrict__ cnt,
    const float* __restrict__ S_acc, const float* __restrict__ CL,
    float* __restrict__ f_out, float* __restrict__ uloss_out)
{
    __shared__ float red[128];
    const int tid = threadIdx.x;
    float ds = 0.0f;
    for (int p = tid; p < KC * KC; p += 128) {
        int a = p / KC, b = p % KC;
        float dd = 0.0f;
        for (int d = 0; d < NZD; ++d) {
            float df = CL[a * NZD + d] - CL[b * NZD + d];
            dd += df * df;
        }
        ds += dd;
    }
    red[tid] = ds;
    __syncthreads();
    for (int s = 64; s > 0; s >>= 1) {
        if (tid < s) red[tid] += red[tid + s];
        __syncthreads();
    }
    if (tid == 0) {
        float mean_d = red[0] / (float)(KC * KC - KC);
        *uloss_out = 0.01f / mean_d;

        float S = *S_acc;
        float um = 0.0f;
        for (int j = 0; j < KC; ++j) um += u_raw[j];
        um /= (float)KC;
        float uv = 0.0f;
        for (int j = 0; j < KC; ++j) { float d = u_raw[j] - um; uv += d * d; }
        float ustd = sqrtf(uv / (float)(KC - 1));

        float vvals[KC];
        float vm = 0.0f;
        for (int j = 0; j < KC; ++j) {
            float c = (cnt[j] > 0) ? (float)cnt[j] : 1.0f;
            vvals[j] = sqrtf(c) * S;
            vm += vvals[j];
        }
        vm /= (float)KC;
        float vv = 0.0f;
        for (int j = 0; j < KC; ++j) { float d = vvals[j] - vm; vv += d * d; }
        float vstd = sqrtf(vv / (float)(KC - 1));

        float umin = 1e30f, vmin = 1e30f;
        for (int j = 0; j < KC; ++j) {
            float un = (u_raw[j] - um) / ustd;
            float vn = (vvals[j] - vm) / vstd;
            umin = fminf(umin, un);
            vmin = fminf(vmin, vn);
        }
        for (int j = 0; j < KC; ++j) {
            float un = (u_raw[j] - um) / ustd - umin + 0.001f;
            float vn = (vvals[j] - vm) / vstd - vmin + 0.001f;
            f_out[j] = un + vn + CCONST;
        }
    }
}

// ---------------- KL divergence pass ----------------
__global__ __launch_bounds__(256) void kl_kernel(
    const float* __restrict__ Qm, const float* __restrict__ f,
    float* __restrict__ kl_acc, int Npts)
{
    __shared__ float fl[KC];
    __shared__ float red[256];
    const int tid = threadIdx.x;
    for (int i = tid; i < KC; i += 256) fl[i] = f[i];
    __syncthreads();

    float ksum = 0.0f;
    const int i = blockIdx.x * 256 + tid;
    if (i < Npts) {
        const float* q = Qm + (size_t)i * KC;
        float ws = 0.0f;
        for (int j = 0; j < KC; ++j) { float qq = q[j]; ws += qq * qq / fl[j]; }
        float invws = 1.0f / ws;
        for (int j = 0; j < KC; ++j) {
            float qq = q[j];
            float p = (qq * qq / fl[j]) * invws;   // p = p0
            ksum += p * (logf(p) - logf(qq));
        }
    }
    red[tid] = ksum;
    __syncthreads();
    for (int s = 128; s > 0; s >>= 1) {
        if (tid < s) red[tid] += red[tid + s];
        __syncthreads();
    }
    if (tid == 0) atomicAdd(kl_acc, red[0]);
}

// ---------------- small helpers ----------------
__global__ void zero_kernel(float* u, int* cnt, float* scal) {
    int i = threadIdx.x;
    if (i < KC) { u[i] = 0.0f; cnt[i] = 0; }
    if (i < 16) scal[i] = 0.0f;
}
__global__ void pred_out_kernel(const int* __restrict__ PRED, float* __restrict__ out, int Npts) {
    int i = blockIdx.x * 256 + threadIdx.x;
    if (i < Npts) out[i] = (float)PRED[i];
}
__global__ void loss_kernel(const float* kl_acc, const float* re_acc,
                            const float* uloss, float* out, int Npts) {
    if (threadIdx.x == 0 && blockIdx.x == 0) {
        float kl = (*kl_acc) / ((float)Npts * (float)KC) * 0.01f;
        float re = (*re_acc) / ((float)Npts * 1024.0f);
        out[Npts] = kl + re + (*uloss);
    }
}

// ---------------- host launcher ----------------
extern "C" void kernel_launch(void* const* d_in, const int* in_sizes, int n_in,
                              void* d_out, int out_size, void* d_ws, size_t ws_size,
                              hipStream_t stream) {
    const float* x   = (const float*)d_in[0];
    const float* t1  = (const float*)d_in[1];
    const float* CL  = (const float*)d_in[2];
    const float* We1 = (const float*)d_in[3];
    const float* be1 = (const float*)d_in[4];
    const float* We2 = (const float*)d_in[5];
    const float* be2 = (const float*)d_in[6];
    const float* We3 = (const float*)d_in[7];
    const float* be3 = (const float*)d_in[8];
    const float* Wz  = (const float*)d_in[9];
    const float* bz  = (const float*)d_in[10];
    const float* Wd1 = (const float*)d_in[11];
    const float* bd1 = (const float*)d_in[12];
    const float* Wd2 = (const float*)d_in[13];
    const float* bd2 = (const float*)d_in[14];
    const float* Wd3 = (const float*)d_in[15];
    const float* bd3 = (const float*)d_in[16];
    const float* Wxb = (const float*)d_in[17];
    const float* bxb = (const float*)d_in[18];
    float* out = (float*)d_out;

    const size_t N = N_PTS;
    // f16 region
    f16_t* h = (f16_t*)d_ws;
    f16_t* Xh   = h; h += N * 1024;
    f16_t* P16  = h; h += N * 2048;     // h3, then d1
    f16_t* Q1h  = h; h += N * 512;      // h1, then d2
    f16_t* Q2h  = h; h += N * 512;      // h2, then d3
    f16_t* Zh   = h; h += N * 32;       // z (f16 copy for decoder)
    f16_t* Wt1  = h; h += 512 * 1024;
    f16_t* Wt2  = h; h += 512 * 512;
    f16_t* Wt3  = h; h += 2048 * 512;
    f16_t* Wtz  = h; h += 128 * 2048;   // zero-padded 32 -> 128 rows
    f16_t* Wtd1 = h; h += 2048 * 32;
    f16_t* Wtd2 = h; h += 512 * 2048;
    f16_t* Wtd3 = h; h += 512 * 512;
    f16_t* Wtxb = h; h += 1024 * 512;
    // f32 region (16B aligned)
    size_t off16 = (size_t)(h - (f16_t*)d_ws);
    off16 = (off16 + 7) & ~(size_t)7;
    float* Zb    = (float*)((f16_t*)d_ws + off16);   // N x 32 (f32 z for clustering)
    float* Qm    = Zb + N * 32;                      // N x 100
    int*   PRED  = (int*)(Qm + N * 100);             // N
    float* u_raw = (float*)(PRED + N);               // 128
    int*   cnt   = (int*)(u_raw + 128);              // 128
    float* scal  = (float*)(cnt + 128);              // [0]=S [1]=re [2]=kl [3]=uloss
    float* f_out = scal + 16;                        // 128

    const int GM = (int)(N / BM);                    // 232
    dim3 blk(256);

    zero_kernel<<<1, 128, 0, stream>>>(u_raw, cnt, scal);

    // ---- prep: x -> f16, weights -> transposed f16 ----
    {
        int nx = (int)(N * 1024);
        cvt_f16_kernel<<<(nx + 255) / 256, 256, 0, stream>>>(x, Xh, nx);
        dim3 tb(32, 8);
        auto tc = [&](const float* W, f16_t* Wt, int K, int M, int Mpad) {
            dim3 g((K + 31) / 32, (Mpad + 31) / 32);
            transpose_cvt_kernel<<<g, tb, 0, stream>>>(W, Wt, K, M, Mpad);
        };
        tc(We1, Wt1, 1024, 512, 512);
        tc(We2, Wt2, 512, 512, 512);
        tc(We3, Wt3, 512, 2048, 2048);
        tc(Wz,  Wtz, 2048, 32, 128);
        tc(Wd1, Wtd1, 32, 2048, 2048);
        tc(Wd2, Wtd2, 2048, 512, 512);
        tc(Wd3, Wtd3, 512, 512, 512);
        tc(Wxb, Wtxb, 512, 1024, 1024);
    }

    // ---- encoder ----
    gemm_kernel<1, false><<<dim3(GM, 512 / BN),  blk, 0, stream>>>(Xh,  Wt1, be1, Q1h, nullptr, nullptr, nullptr, 1024, 512);
    gemm_kernel<1, false><<<dim3(GM, 512 / BN),  blk, 0, stream>>>(Q1h, Wt2, be2, Q2h, nullptr, nullptr, nullptr, 512,  512);
    gemm_kernel<1, false><<<dim3(GM, 2048 / BN), blk, 0, stream>>>(Q2h, Wt3, be3, P16, nullptr, nullptr, nullptr, 512,  2048);
    gemm_kernel<0, true ><<<dim3(GM, 1),         blk, 0, stream>>>(P16, Wtz, bz,  Zh,  Zb,      nullptr, nullptr, 2048, 32);
    // ---- decoder ----
    gemm_kernel<1, false><<<dim3(GM, 2048 / BN), blk, 0, stream>>>(Zh,  Wtd1, bd1, P16, nullptr, nullptr, nullptr, 32,   2048);
    gemm_kernel<1, false><<<dim3(GM, 512 / BN),  blk, 0, stream>>>(P16, Wtd2, bd2, Q1h, nullptr, nullptr, nullptr, 2048, 512);
    gemm_kernel<1, false><<<dim3(GM, 512 / BN),  blk, 0, stream>>>(Q1h, Wtd3, bd3, Q2h, nullptr, nullptr, nullptr, 512,  512);
    // reconstruction loss fused into last GEMM (no x_bar buffer)
    gemm_kernel<2, false><<<dim3(GM, 1024 / BN), blk, 0, stream>>>(Q2h, Wtxb, bxb, nullptr, nullptr, x, &scal[1], 512, 1024);

    // ---- clustering ----
    cluster_kernel<<<512, 256, 0, stream>>>(Zb, t1, CL, Qm, PRED, u_raw, cnt, &scal[0], (int)N);
    finalize_kernel<<<1, 128, 0, stream>>>(u_raw, cnt, &scal[0], CL, f_out, &scal[3]);
    kl_kernel<<<(int)((N + 255) / 256), 256, 0, stream>>>(Qm, f_out, &scal[2], (int)N);

    // ---- outputs ----
    pred_out_kernel<<<(int)((N + 255) / 256), 256, 0, stream>>>(PRED, out, (int)N);
    loss_kernel<<<1, 1, 0, stream>>>(&scal[2], &scal[1], &scal[3], out, (int)N);
}
